// torch_model_5858335392187
// MI455X (gfx1250) — matrix operand order repaired and checked by reading, not compiled, not run on hardware
//
#include <hip/hip_runtime.h>

typedef __attribute__((ext_vector_type(16))) __bf16 v16bf;
typedef __attribute__((ext_vector_type(8)))  float  v8f;
typedef __attribute__((ext_vector_type(4)))  unsigned short v4u;
typedef __attribute__((ext_vector_type(8)))  unsigned short v8u;
typedef __attribute__((ext_vector_type(16))) unsigned short v16u;

#define DIMS 64
#define KNN 21
#define GN_MAX 128
#define CAND_PER_WG 42                /* 2 halves * 21 */
#define NC_MAX (GN_MAX * CAND_PER_WG) /* 5376 */
#define ROWP 72                       /* padded row stride (bf16 elems) for staged tiles */

// ---------------------------------------------------------------------------
// Stage 1: bf16 WMMA (hi/lo split -> fp32-accurate) scoring with async-to-LDS
// tile staging, block-cooperative conversion, per-lane top-21 lists.
// ---------------------------------------------------------------------------
__global__ __launch_bounds__(256) void knn_stage1(
    const float* __restrict__ xq, const float* __restrict__ xb,
    float* __restrict__ cs, int* __restrict__ ci,
    int N, int Gn, int chunk)
{
  __shared__ float s_sc[256 * KNN];
  __shared__ int   s_ix[256 * KNN];
  __shared__ __align__(16) unsigned short s_hi[2][16 * ROWP];
  __shared__ __align__(16) unsigned short s_lo[2][16 * ROWP];
  __shared__ __align__(16) float          s_raw[2][16 * DIMS];

  const int tid  = threadIdx.x;
  const int wave = tid >> 5;
  const int lane = tid & 31;
  const int col  = lane & 15;   // WMMA N column = this lane's query; also A row it maps to
  const int half = lane >> 4;   // which 8-row half of C this lane holds
  const int gn   = blockIdx.x;
  const int q    = (int)blockIdx.y * 128 + wave * 16 + col;

  float* mysc = &s_sc[tid * KNN];
  int*   myix = &s_ix[tid * KNN];
#pragma unroll
  for (int j = 0; j < KNN; ++j) { mysc[j] = -3.402823466e38f; myix[j] = 0; }
  float thr    = -3.402823466e38f;
  int   minpos = 0;

  // Resident B-operand (queries): hi/lo bf16 split, two K-chunks of 32.
  v16bf hiB[2], loB[2];
  {
    const float* qp = xq + (size_t)q * DIMS + half * 16;
#pragma unroll
    for (int c = 0; c < 2; ++c) {
#pragma unroll
      for (int i = 0; i < 16; ++i) {
        float  x = qp[c * 32 + i];
        __bf16 h = (__bf16)x;
        hiB[c][i] = h;
        loB[c][i] = (__bf16)(x - (float)h);
      }
    }
  }

  const long long nstart = (long long)gn * chunk;
  long long nrem = (long long)N - nstart;
  if (nrem < 0) nrem = 0;
  const long long ncnt = nrem < chunk ? nrem : (long long)chunk;
  const int nt = (int)((ncnt + 15) >> 4);

  // --- helpers -------------------------------------------------------------
  auto issue_async = [&](int tt, int b) {
    // Each thread DMAs 16B of the contiguous 4KB tile straight into LDS.
    long long e = (nstart + (long long)tt * 16) * DIMS + (long long)tid * 4;
    const long long emax = (long long)N * DIMS - 4;
    if (e > emax) e = emax;  // clamp; bogus rows filtered at harvest
    const unsigned dst = (unsigned)(size_t)&s_raw[b][tid * 4];
    const unsigned long long src = (unsigned long long)(size_t)(xb + e);
    asm volatile("global_load_async_to_lds_b128 %0, %1, off"
                 :: "v"(dst), "v"(src) : "memory");
  };
  auto wait_async = [&]() {
    asm volatile("s_wait_asynccnt 0x0" ::: "memory");
  };
  auto convert = [&](int b) {
    // 256 threads x 4 elems: f32 tile -> bf16 hi/lo tiles (row-padded).
    const int e0  = tid * 4;
    const int row = e0 >> 6, k = e0 & 63;
    const float4 x4 = *(const float4*)&s_raw[b][e0];
    const float  xs[4] = {x4.x, x4.y, x4.z, x4.w};
    v4u hv, lv;
#pragma unroll
    for (int i = 0; i < 4; ++i) {
      __bf16 h = (__bf16)xs[i];
      hv[i] = __builtin_bit_cast(unsigned short, h);
      lv[i] = __builtin_bit_cast(unsigned short, (__bf16)(xs[i] - (float)h));
    }
    *(v4u*)&s_hi[b][row * ROWP + k] = hv;
    *(v4u*)&s_lo[b][row * ROWP + k] = lv;
  };
  // -------------------------------------------------------------------------

  if (nt > 0) {
    issue_async(0, 0);
    wait_async();
    __syncthreads();
    convert(0);
    if (nt > 1) issue_async(1, 1);
    __syncthreads();
  }

  for (int t = 0; t < nt; ++t) {
    const int cur = t & 1;
    const long long nb = nstart + (long long)t * 16;

    v8f acc = {};
    const int rb = col * ROWP;
#pragma unroll
    for (int c = 0; c < 2; ++c) {
      // A element i -> K = 32c + 16*half + i: contiguous 16.
      const int o1 = c * 32 + half * 16;
      v8u h0 = *(const v8u*)&s_hi[cur][rb + o1];
      v8u h1 = *(const v8u*)&s_hi[cur][rb + o1 + 8];
      v8u l0 = *(const v8u*)&s_lo[cur][rb + o1];
      v8u l1 = *(const v8u*)&s_lo[cur][rb + o1 + 8];
      v16bf hiA = __builtin_bit_cast(v16bf, __builtin_shufflevector(
          h0, h1, 0, 1, 2, 3, 4, 5, 6, 7, 8, 9, 10, 11, 12, 13, 14, 15));
      v16bf loA = __builtin_bit_cast(v16bf, __builtin_shufflevector(
          l0, l1, 0, 1, 2, 3, 4, 5, 6, 7, 8, 9, 10, 11, 12, 13, 14, 15));
      // D = A*B + C ; fp32-accurate via hi*hi + hi*lo + lo*hi
      acc = __builtin_amdgcn_wmma_f32_16x16x32_bf16(false, hiA, false, hiB[c],
                                                    (short)0, acc, false, false);
      acc = __builtin_amdgcn_wmma_f32_16x16x32_bf16(false, hiA, false, loB[c],
                                                    (short)0, acc, false, false);
      acc = __builtin_amdgcn_wmma_f32_16x16x32_bf16(false, loA, false, hiB[c],
                                                    (short)0, acc, false, false);
    }

    // Harvest: C vgpr v -> db row nb + 8*half + v, query = col (this lane).
#pragma unroll
    for (int v = 0; v < 8; ++v) {
      const long long idx = nb + half * 8 + v;
      const float s = acc[v];
      if (idx < N && s > thr) {
        mysc[minpos] = s;
        myix[minpos] = (int)idx;
        float m = mysc[0]; int mp = 0;
#pragma unroll
        for (int j = 1; j < KNN; ++j) {
          float sj = mysc[j];
          if (sj < m) { m = sj; mp = j; }
        }
        thr = m; minpos = mp;
      }
    }

    if (t + 1 < nt) {
      wait_async();          // tile t+1 raw data arrived (per-wave portion)
      __syncthreads();       // all waves' async portions visible
      convert((t + 1) & 1);
      if (t + 2 < nt) issue_async(t + 2, t & 1);
      __syncthreads();       // converted tile visible before next compute
    }
  }

  // Emit this lane's 21 candidates for query q.
  const size_t base = (size_t)q * ((size_t)Gn * CAND_PER_WG) +
                      (size_t)gn * CAND_PER_WG + (size_t)half * KNN;
#pragma unroll
  for (int j = 0; j < KNN; ++j) { cs[base + j] = mysc[j]; ci[base + j] = myix[j]; }
}

// ---------------------------------------------------------------------------
// Stage 2: per query, merge Gn*42 candidates -> global top-21 (descending,
// ties broken by lower index like jax.lax.top_k).
// ---------------------------------------------------------------------------
__global__ __launch_bounds__(256) void knn_stage2(
    const float* __restrict__ cs, const int* __restrict__ ci,
    int* __restrict__ out, int nc)
{
  __shared__ float ls[NC_MAX];
  __shared__ int   li[NC_MAX];
  __shared__ float rs[256];
  __shared__ int   ri[256];
  __shared__ int   rp[256];

  const int q = blockIdx.x;
  const int tid = threadIdx.x;
  const float* csq = cs + (size_t)q * nc;
  const int*   ciq = ci + (size_t)q * nc;
  for (int i = tid; i < nc; i += 256) { ls[i] = csq[i]; li[i] = ciq[i]; }
  __syncthreads();

  for (int r = 0; r < KNN; ++r) {
    float b = -__builtin_inff(); int bi = 0x7fffffff; int bp = -1;
    for (int i = tid; i < nc; i += 256) {
      const float s = ls[i]; const int ix = li[i];
      if (s > b || (s == b && ix < bi)) { b = s; bi = ix; bp = i; }
    }
    rs[tid] = b; ri[tid] = bi; rp[tid] = bp;
    __syncthreads();
    if (tid == 0) {
      float bb = rs[0]; int bbi = ri[0]; int bbp = rp[0];
      for (int u = 1; u < 256; ++u) {
        if (rs[u] > bb || (rs[u] == bb && ri[u] < bbi)) {
          bb = rs[u]; bbi = ri[u]; bbp = rp[u];
        }
      }
      out[q * KNN + r] = bbi;
      if (bbp >= 0) ls[bbp] = -3.402823466e38f;  // remove winner
    }
    __syncthreads();
  }
}

// ---------------------------------------------------------------------------
extern "C" void kernel_launch(void* const* d_in, const int* in_sizes, int n_in,
                              void* d_out, int out_size, void* d_ws, size_t ws_size,
                              hipStream_t stream) {
  const float* xq = (const float*)d_in[0];
  const float* xb = (const float*)d_in[1];
  const int B = in_sizes[0] / DIMS;   // 256
  const int N = in_sizes[1] / DIMS;   // 1,000,000
  int qblocks = B / 128; if (qblocks < 1) qblocks = 1;

  // Workspace: B * Gn * 42 candidates, (f32 score + i32 idx) each.
  const size_t perG = (size_t)B * CAND_PER_WG * 8;
  int Gn = GN_MAX;
  if ((size_t)Gn * perG > ws_size) Gn = (int)(ws_size / perG);
  if (Gn < 1) Gn = 1;
  if (Gn > GN_MAX) Gn = GN_MAX;
  const int chunk = (N + Gn - 1) / Gn;

  float* cs = (float*)d_ws;
  int*   ci = (int*)((char*)d_ws + (size_t)B * Gn * CAND_PER_WG * sizeof(float));

  dim3 g1(Gn, qblocks);
  knn_stage1<<<g1, 256, 0, stream>>>(xq, xb, cs, ci, N, Gn, chunk);
  knn_stage2<<<B, 256, 0, stream>>>(cs, ci, (int*)d_out, Gn * CAND_PER_WG);
}
